// EGULayer_5540507812335
// MI455X (gfx1250) — matrix operand order repaired and checked by reading, not compiled, not run on hardware
//
#include <hip/hip_runtime.h>

#define B_DIM 16
#define S_DIM 4096
#define IN_DIM 256
#define H_DIM 256
#define LN_EPS 1e-5f

typedef __attribute__((ext_vector_type(16))) __bf16 v16bf;
typedef __attribute__((ext_vector_type(8)))  float v8f;
typedef __attribute__((ext_vector_type(8)))  unsigned int v8u;

union ABf {
    v16bf v;
    __bf16 e[16];
    uint4 u4[2];
    v8u u8;
};

__device__ __forceinline__ float fast_sigmoid(float z) {
    // 1 / (1 + e^-z)  -> v_exp_f32 + v_rcp_f32
    return __builtin_amdgcn_rcpf(1.0f + __expf(-z));
}

// ---------------------------------------------------------------------------
// Kernel 1: x_silu = silu(x @ Wx^T + bx), written into d_out[B*S, H] (f32).
// Grid: (B*S)/16 blocks of 512 threads (16 waves). Each wave: one 16x16 tile.
// ---------------------------------------------------------------------------
__global__ __launch_bounds__(512) void egu_xproj(const float* __restrict__ x,
                                                 const float* __restrict__ Wx,
                                                 const float* __restrict__ bx,
                                                 float* __restrict__ out) {
    __shared__ __attribute__((aligned(16))) __bf16 xb[16][264];  // 16B-pad rows

    const int tid  = threadIdx.x;
    const int lane = tid & 31;
    const int wave = tid >> 5;
    const int row0 = blockIdx.x * 16;

    // ---- Preload B-matrix (Wx) into bf16 registers, B layout with contiguous k (same k order as A):
    // lane (0..15 | 16..31): n = 16*wave + (lane&15), kb = 0 | 16,
    // element j of k-step kk is Wx[n][32*kk + kb + j].
    const int nloc = lane & 15;
    const int n    = wave * 16 + nloc;
    const int kbB  = (lane < 16) ? 0 : 16;
    ABf Breg[8];
#pragma unroll
    for (int kk = 0; kk < 8; ++kk) {
        const float* wp = Wx + (size_t)n * IN_DIM + kk * 32 + kbB;
#pragma unroll
        for (int j = 0; j < 16; ++j) Breg[kk].e[j] = (__bf16)wp[j];
        // Pin converted bf16 values in VGPRs (block f32->bf16 remat in loop).
        asm volatile("" : "+v"(Breg[kk].u8));
    }

    // ---- Cooperative load: 16x256 f32 tile of x -> bf16 LDS
    {
        const int r  = tid >> 5;          // 16 rows
        const int c0 = (tid & 31) * 8;    // 8 elems / lane
        const float* src = x + (size_t)(row0 + r) * IN_DIM + c0;
        const float4 f0 = *(const float4*)(src);
        const float4 f1 = *(const float4*)(src + 4);
        __bf16* dst = &xb[r][c0];
        dst[0] = (__bf16)f0.x; dst[1] = (__bf16)f0.y;
        dst[2] = (__bf16)f0.z; dst[3] = (__bf16)f0.w;
        dst[4] = (__bf16)f1.x; dst[5] = (__bf16)f1.y;
        dst[6] = (__bf16)f1.z; dst[7] = (__bf16)f1.w;
    }
    __syncthreads();

    // ---- A layout: lane m = lane&15, kb = 0|16; elements = k kb..kb+15
    const int mA  = lane & 15;
    const int kbA = (lane < 16) ? 0 : 16;
    v8f acc = {};
#pragma unroll
    for (int kk = 0; kk < 8; ++kk) {
        ABf A;
        const __bf16* ap = &xb[mA][kk * 32 + kbA];
        A.u4[0] = *(const uint4*)(ap);        // ds_load_b128
        A.u4[1] = *(const uint4*)(ap + 8);    // ds_load_b128
        acc = __builtin_amdgcn_wmma_f32_16x16x32_bf16(
            false, A.v, false, Breg[kk].v, (short)0, acc, false, false);
    }

    // ---- Epilogue: silu(acc + bx), C/D layout: VGPR r -> M = r + 8*(lane>=16)
    const float bias = bx[n];
    const int mBase  = (lane < 16) ? 0 : 8;
#pragma unroll
    for (int r = 0; r < 8; ++r) {
        const int m = mBase + r;
        const float z = acc[r] + bias;
        out[(size_t)(row0 + m) * H_DIM + n] = z * fast_sigmoid(z);
    }
}

// ---------------------------------------------------------------------------
// Kernel 2: persistent sequential scan. One workgroup (512 thr = 16 waves).
// h (16x256) lives in an LDS double buffer (bf16); Wh pinned in VGPRs (bf16).
// Per step per wave: 8x v_wmma_f32_16x16x32_bf16 + gating; 1 barrier/step.
// Reads xs_t from d_out, overwrites the slot with h_t.
// ---------------------------------------------------------------------------
__global__ __launch_bounds__(512) void egu_scan(const float* __restrict__ Wh,
                                                const float* __restrict__ bh,
                                                float* __restrict__ out) {
    __shared__ __attribute__((aligned(16))) __bf16 hbuf[2][16][264];

    const int tid   = threadIdx.x;
    const int lane  = tid & 31;
    const int wave  = tid >> 5;
    const int nloc  = lane & 15;
    const int n     = wave * 16 + nloc;
    const int kbB   = (lane < 16) ? 0 : 16;
    const int kbA   = (lane < 16) ? 0 : 16;
    const int mA    = lane & 15;
    const int mBase = (lane < 16) ? 0 : 8;

    // ---- Pin Wh (bf16, B layout) in registers: 64 VGPRs/lane, converted ONCE.
    ABf Breg[8];
#pragma unroll
    for (int kk = 0; kk < 8; ++kk) {
        const float* wp = Wh + (size_t)n * H_DIM + kk * 32 + kbB;
#pragma unroll
        for (int j = 0; j < 16; ++j) Breg[kk].e[j] = (__bf16)wp[j];
        // Opaque def: forbid per-iteration f32->bf16 rematerialization.
        asm volatile("" : "+v"(Breg[kk].u8));
    }
    const float bias = bh[n];

    // ---- h0 = 0
    for (int i = tid; i < 16 * 264; i += 512) ((__bf16*)hbuf[0])[i] = (__bf16)0.0f;
    __syncthreads();

    for (int t = 0; t < S_DIM; ++t) {
        const int cur = t & 1;
        const int nxt = cur ^ 1;

        // Prefetch xs_t early so global loads overlap the WMMA chain.
        float xs[8];
#pragma unroll
        for (int r = 0; r < 8; ++r) {
            const int m = mBase + r;  // m == batch index
            xs[r] = out[((size_t)m * S_DIM + t) * H_DIM + n];
        }

        // h_silu pre-act = h @ Wh^T  (A = h from LDS, B = Wh from regs)
        v8f acc = {};
#pragma unroll
        for (int kk = 0; kk < 8; ++kk) {
            ABf A;
            const __bf16* ap = &hbuf[cur][mA][kk * 32 + kbA];
            A.u4[0] = *(const uint4*)(ap);
            A.u4[1] = *(const uint4*)(ap + 8);
            acc = __builtin_amdgcn_wmma_f32_16x16x32_bf16(
                false, A.v, false, Breg[kk].v, (short)0, acc, false, false);
        }

        // Gating + state update; stage h_{t+1} into the other LDS buffer.
#pragma unroll
        for (int r = 0; r < 8; ++r) {
            const int m = mBase + r;
            const float z  = acc[r] + bias;
            const float hs = z * fast_sigmoid(z);                 // silu
            const float g  = fast_sigmoid(xs[r] + hs);
            const float hn = g * xs[r] + (1.0f - g) * hs;
            out[((size_t)m * S_DIM + t) * H_DIM + n] = hn;        // emit h_t
            hbuf[nxt][m][n] = (__bf16)hn;                         // next A
        }
        __syncthreads();  // all writes to hbuf[nxt] visible before next step
    }
}

// ---------------------------------------------------------------------------
// Kernel 3: in-place LayerNorm over each H=256 row of d_out. One wave / row.
// ---------------------------------------------------------------------------
__global__ __launch_bounds__(256) void egu_ln(float* __restrict__ out,
                                              const float* __restrict__ gamma,
                                              const float* __restrict__ beta) {
    const int lane = threadIdx.x & 31;
    const int wrow = blockIdx.x * (256 / 32) + (threadIdx.x >> 5);
    float* row = out + (size_t)wrow * H_DIM;
    const int c0 = lane * 8;

    float4 a = *(const float4*)(row + c0);
    float4 b = *(const float4*)(row + c0 + 4);

    float s = a.x + a.y + a.z + a.w + b.x + b.y + b.z + b.w;
    float q = a.x * a.x + a.y * a.y + a.z * a.z + a.w * a.w +
              b.x * b.x + b.y * b.y + b.z * b.z + b.w * b.w;
#pragma unroll
    for (int m = 16; m >= 1; m >>= 1) {
        s += __shfl_xor(s, m, 32);
        q += __shfl_xor(q, m, 32);
    }
    const float mean = s * (1.0f / 256.0f);
    const float var  = q * (1.0f / 256.0f) - mean * mean;
    const float rs   = rsqrtf(var + LN_EPS);

    const float4 g0 = *(const float4*)(gamma + c0);
    const float4 g1 = *(const float4*)(gamma + c0 + 4);
    const float4 p0 = *(const float4*)(beta + c0);
    const float4 p1 = *(const float4*)(beta + c0 + 4);

    a.x = (a.x - mean) * rs * g0.x + p0.x;
    a.y = (a.y - mean) * rs * g0.y + p0.y;
    a.z = (a.z - mean) * rs * g0.z + p0.z;
    a.w = (a.w - mean) * rs * g0.w + p0.w;
    b.x = (b.x - mean) * rs * g1.x + p1.x;
    b.y = (b.y - mean) * rs * g1.y + p1.y;
    b.z = (b.z - mean) * rs * g1.z + p1.z;
    b.w = (b.w - mean) * rs * g1.w + p1.w;

    *(float4*)(row + c0)     = a;
    *(float4*)(row + c0 + 4) = b;
}

// ---------------------------------------------------------------------------
extern "C" void kernel_launch(void* const* d_in, const int* in_sizes, int n_in,
                              void* d_out, int out_size, void* d_ws, size_t ws_size,
                              hipStream_t stream) {
    (void)in_sizes; (void)n_in; (void)d_ws; (void)ws_size; (void)out_size;
    const float* x     = (const float*)d_in[0];
    const float* Wx    = (const float*)d_in[1];
    const float* bx    = (const float*)d_in[2];
    const float* Wh    = (const float*)d_in[3];
    const float* bh    = (const float*)d_in[4];
    const float* gamma = (const float*)d_in[5];
    const float* beta  = (const float*)d_in[6];
    float* out = (float*)d_out;

    // 1) x-path projection + SiLU -> d_out (staging, bandwidth-bound)
    egu_xproj<<<(B_DIM * S_DIM) / 16, 512, 0, stream>>>(x, Wx, bx, out);
    // 2) latency-critical recurrence: single persistent workgroup on one WGP
    egu_scan<<<1, 512, 0, stream>>>(Wh, bh, out);
    // 3) in-place LayerNorm (bandwidth-bound)
    egu_ln<<<(B_DIM * S_DIM) / 8, 256, 0, stream>>>(out, gamma, beta);
}
